// MotifPredictor_56667798503475
// MI455X (gfx1250) — matrix operand order repaired and checked by reading, not compiled, not run on hardware
//
#include <hip/hip_runtime.h>

// ---------------------------------------------------------------------------
// MotifPredictor on MI455X (gfx1250, wave32, WMMA + async/TDM data movers)
//
//   k0  : f32->bf16 weight converts (+ W_rel zero-padded to 64 rows)
//   k1  : edge_rep GEMM (WMMA bf16) -> head/tail bf16 tables in ws
//   k2  : fused  gather -> GEMM2 (x W_post_cat^T) -> *union -> GEMM3 (x W_rel^T)
//         -> + b_rel + freq_bias gather -> out  (prod_rep never hits HBM)
//
// Data movement: B tiles via Tensor Data Mover (tensor_load_to_lds, TENSORcnt,
// double-buffered so the DMA overlaps the WMMA chain); A/union/W_rel staging
// via GLOBAL_LOAD_ASYNC_TO_LDS_B128 (ASYNCcnt).  Guarded by __has_builtin with
// synchronous fallbacks.
// ---------------------------------------------------------------------------

typedef __attribute__((ext_vector_type(16))) __bf16 v16bf;
typedef __attribute__((ext_vector_type(8)))  __bf16 v8bf;
typedef __attribute__((ext_vector_type(8)))  float  v8f;
typedef __attribute__((ext_vector_type(4)))  unsigned int u32x4;
typedef __attribute__((ext_vector_type(8)))  int i32x8;
typedef __attribute__((ext_vector_type(4)))  int i32x4;

// GCC-style vector pointee types for the async-load builtin's pointer params
// (diagnostic showed: "__attribute__((__vector_size__(4*sizeof(int)))) int *").
typedef int v4i_vs __attribute__((vector_size(16)));
typedef v4i_vs __attribute__((address_space(1))) as1_v4i;  // global
typedef v4i_vs __attribute__((address_space(3))) as3_v4i;  // LDS

#if defined(__HIP_DEVICE_COMPILE__) &&                                        \
    __has_builtin(__builtin_amdgcn_global_load_async_to_lds_b128) &&          \
    __has_builtin(__builtin_amdgcn_s_wait_asynccnt)
#define HAVE_ASYNC 1
#else
#define HAVE_ASYNC 0
#endif
#if defined(__HIP_DEVICE_COMPILE__) &&                                        \
    __has_builtin(__builtin_amdgcn_tensor_load_to_lds) &&                     \
    __has_builtin(__builtin_amdgcn_s_wait_tensorcnt)
#define HAVE_TDM 1
#else
#define HAVE_TDM 0
#endif

// 16B global -> LDS copy: async (no VGPR round trip) when available.
static __device__ __forceinline__ void cp16(const void* g, void* l) {
#if HAVE_ASYNC
  __builtin_amdgcn_global_load_async_to_lds_b128(
      (as1_v4i*)(unsigned long long)g,
      (as3_v4i*)(unsigned int)(unsigned long long)l,
      0, 0);
#else
  *(uint4*)l = *(const uint4*)g;
#endif
}

static __device__ __forceinline__ void wait_async0() {
#if HAVE_ASYNC
  __builtin_amdgcn_s_wait_asynccnt(0);
#endif
}

static __device__ __forceinline__ void wait_tensor0() {
#if HAVE_TDM
  __builtin_amdgcn_s_wait_tensorcnt(0);
#endif
}

static __device__ __forceinline__ v8f wmma_bf16(v16bf a, v16bf b, v8f c) {
  return __builtin_amdgcn_wmma_f32_16x16x32_bf16(
      false, a, false, b, (short)0, c, false, false);
}

// A-fragment (ISA 7.12.2, 16-bit A 16x32): lane (m=lane&15, g=lane>>4) holds
// K = k0 + g*8 + [0..7]  and  K = k0 + 16 + g*8 + [0..7]  of row m.
static __device__ __forceinline__ v16bf ld_frag_a(const __bf16* base, int stride,
                                                  int m, int g, int k0) {
  const __bf16* p = base + m * stride + k0 + g * 8;
  v8bf lo = *(const v8bf*)p;
  v8bf hi = *(const v8bf*)(p + 16);
  v16bf r;
  ((v8bf*)&r)[0] = lo;
  ((v8bf*)&r)[1] = hi;
  return r;
}

// B-fragment (32x16 KxN): lane (n=lane&15, g=lane>>4) holds K = k0+g*8+[0..7] and k0+16+g*8+[0..7]
// of column n.  B stored as rows of W (K contiguous).
static __device__ __forceinline__ v16bf ld_frag_b(const __bf16* base, int stride,
                                                  int n, int g, int k0) {
  const __bf16* p = base + n * stride + k0 + g * 8;
  v8bf lo = *(const v8bf*)p;
  v8bf hi = *(const v8bf*)(p + 16);
  v16bf r;
  ((v8bf*)&r)[0] = lo;
  ((v8bf*)&r)[1] = hi;
  return r;
}

// ---------------------------------------------------------------------------
#define N_OBJ   16384
#define N_REL   32768
#define HID     512
#define POOL    4096
#define NRELCLS 51
#define NOBJCLS 151

// ---------------------------------------------------------------------------
__global__ void cvt_f32_bf16(const float* __restrict__ src, __bf16* __restrict__ dst, int n) {
  int i = blockIdx.x * blockDim.x + threadIdx.x;
  if (i < n) dst[i] = (__bf16)src[i];
}

__global__ void pad_wrel(const float* __restrict__ wrel, __bf16* __restrict__ dst) {
  int i = blockIdx.x * blockDim.x + threadIdx.x;  // 64*4096
  int c = i >> 12, p = i & 4095;
  float v = (c < NRELCLS) ? wrel[c * POOL + p] : 0.0f;
  dst[i] = (__bf16)v;
}

// ---------------------------------------------------------------------------
// k1: edge_rep = edge_ctx @ W_post_emb^T + b ; split to head/tail bf16
#define G1_AS 520

__global__ void gemm1_edge(const float* __restrict__ edge_ctx,
                           const __bf16* __restrict__ wemb,
                           const float* __restrict__ bemb,
                           __bf16* __restrict__ head, __bf16* __restrict__ tail) {
  extern __shared__ char smem[];
  __bf16* A = (__bf16*)smem;
  __bf16* B = A + 64 * G1_AS;

  const int tid = threadIdx.x;
  const int r0 = blockIdx.x * 64, n0 = blockIdx.y * 64;
  const int row = tid >> 2, q = tid & 3;

  // stage B (bf16, async copy) : 128 per thread
  {
    const __bf16* s = wemb + (size_t)(n0 + row) * HID + q * 128;
    __bf16* d = B + row * G1_AS + q * 128;
#pragma unroll
    for (int j = 0; j < 128; j += 8) cp16(s + j, d + j);
  }
  // stage A (f32 -> bf16 convert) : 128 cols per thread
  {
    const float* s = edge_ctx + (size_t)(r0 + row) * HID + q * 128;
    __bf16* d = A + row * G1_AS + q * 128;
#pragma unroll 4
    for (int j = 0; j < 128; j += 4) {
      float4 f = *(const float4*)(s + j);
      d[j + 0] = (__bf16)f.x; d[j + 1] = (__bf16)f.y;
      d[j + 2] = (__bf16)f.z; d[j + 3] = (__bf16)f.w;
    }
  }
  wait_async0();
  __syncthreads();

  const int w = tid >> 5, lane = tid & 31;
  const int rt = w >> 1, nh = w & 1;
  const int m = lane & 15, g = lane >> 4;

  v8f acc[2] = {};
  for (int k0 = 0; k0 < HID; k0 += 32) {
    v16bf a = ld_frag_a(A + rt * 16 * G1_AS, G1_AS, m, g, k0);
#pragma unroll
    for (int c = 0; c < 2; ++c) {
      int ct = nh + 2 * c;
      v16bf b = ld_frag_b(B + ct * 16 * G1_AS, G1_AS, m, g, k0);
      acc[c] = wmma_bf16(a, b, acc[c]);
    }
  }

#pragma unroll
  for (int c = 0; c < 2; ++c) {
    int ct = nh + 2 * c;
    int ncol = n0 + ct * 16 + m;
    float bias = bemb[ncol];
#pragma unroll
    for (int i = 0; i < 8; ++i) {
      int r = r0 + rt * 16 + i + 8 * g;
      __bf16 bv = (__bf16)(acc[c][i] + bias);
      if (ncol < HID) head[(size_t)r * HID + ncol] = bv;
      else            tail[(size_t)r * HID + (ncol - HID)] = bv;
    }
  }
}

// ---------------------------------------------------------------------------
// k2: fused gather + GEMM2 + *union + GEMM3 + bias epilogue.
#define G2_AS 1032  // A stride (1024 + 8)
#define G2_BS 264   // B stride (256 + 8) == 128 DWORD rows + 4 DWORD pad
#define G2_US 68
#define G2_PS 72
#define G2_WS 72

#if HAVE_TDM
// One TDM descriptor loads a full B stage: 64 W_post_cat rows x 256 bf16,
// tensor row stride 1024 elems, LDS-padded to G2_BS (pad 4 DWORDs / 128).
static __device__ __forceinline__ void tdm_issue_B(const __bf16* wc, __bf16* dst,
                                                   int n0, int ks) {
  unsigned long long ga =
      (unsigned long long)(const void*)(wc + (size_t)n0 * 1024 + ks);
  unsigned lds = (unsigned)(unsigned long long)(void*)dst;
  u32x4 g0;
  g0[0] = 1u;                                               // count=1 (user D#)
  g0[1] = lds;                                              // lds_addr (bytes)
  g0[2] = (unsigned)ga;                                     // global_addr lo
  g0[3] = (unsigned)((ga >> 32) & 0x1FFFFFFu) | (2u << 30); // addr[56:32]|type=2
  i32x8 g1;
  g1[0] = (1 << 16)      // data_size = 2 bytes
        | (1 << 20)      // pad_enable
        | (6 << 22)      // pad_interval: 128 DWORDs
        | (3 << 25);     // pad_amount : 4 DWORDs
  g1[1] = (int)(256u << 16);   // tensor_dim0 = 256
  g1[2] = (int)(64u << 16);    // tensor_dim1 = 64
  g1[3] = (int)(256u << 16);   // tile_dim0   = 256
  g1[4] = 64;                  // tile_dim1=64, tile_dim2=0
  g1[5] = 1024;                // tensor_dim0_stride = 1024 elems
  g1[6] = 0;
  g1[7] = 0;
  i32x4 gz4 = {0, 0, 0, 0};
  i32x8 gz8 = {0, 0, 0, 0, 0, 0, 0, 0};
  __builtin_amdgcn_tensor_load_to_lds(g0, g1, gz4, gz4, gz8, 0);
}
#endif

__global__ void gemm2_fused(const __bf16* __restrict__ head, const __bf16* __restrict__ tail,
                            const __bf16* __restrict__ wc,     // [4096][1024] bf16
                            const __bf16* __restrict__ wrelp,  // [64][4096] bf16 padded
                            const float* __restrict__ unionf,  // [32768][4096]
                            const float* __restrict__ bpc,
                            const float* __restrict__ brel,
                            const float* __restrict__ freq,
                            const int* __restrict__ pair_idx,
                            const int* __restrict__ obj_preds,
                            float* __restrict__ out) {
  extern __shared__ char smem[];
  __bf16* A   = (__bf16*)smem;                   // 64 x G2_AS
  __bf16* B   = A + 64 * G2_AS;                  // 2 x 64 x G2_BS (double buf)
  __bf16* P   = B + 2 * 64 * G2_BS;              // 64 x G2_PS
  __bf16* WR  = P + 64 * G2_PS;                  // 64 x G2_WS
  float*  U   = (float*)(WR + 64 * G2_WS);       // 64 x G2_US
  float*  BPC = U + 64 * G2_US;                  // 64
  int*    RB  = (int*)(BPC + 64);                // 64
  int*    HT  = RB + 64;                         // 128

  const int tid = threadIdx.x;
  const int r0 = blockIdx.x * 64;
  const int row = tid >> 2, q = tid & 3;

  if (tid < 64) {
    int h = pair_idx[2 * (r0 + tid) + 0];
    int t = pair_idx[2 * (r0 + tid) + 1];
    HT[2 * tid + 0] = h;
    HT[2 * tid + 1] = t;
    RB[tid] = (obj_preds[h] * NOBJCLS + obj_preds[t]) * NRELCLS;
  }
  __syncthreads();

  // gather A rows via async copies: cols 0..511 <- head[h], 512..1023 <- tail[t]
  {
    int h = HT[2 * row + 0], t = HT[2 * row + 1];
    const __bf16* s = (q < 2) ? head + (size_t)h * HID + q * 256
                              : tail + (size_t)t * HID + (q - 2) * 256;
    __bf16* d = A + row * G2_AS + q * 256;
#pragma unroll
    for (int j = 0; j < 256; j += 8) cp16(s + j, d + j);
  }

  const int w = tid >> 5, lane = tid & 31;
  const int rt = w >> 1, nh = w & 1;
  const int m = lane & 15, g = lane >> 4;

  v8f relacc[2] = {};

  for (int nc = 0; nc < POOL / 64; ++nc) {
    const int n0 = nc * 64;
    __syncthreads();  // prior chunk fully consumed before restaging U/WR/B

    // stage union chunk (async, f32): 16 per thread
    {
      const float* s = unionf + (size_t)(r0 + row) * POOL + n0 + q * 16;
      float* d = U + row * G2_US + q * 16;
#pragma unroll
      for (int j = 0; j < 16; j += 4) cp16(s + j, d + j);
    }
    // stage W_rel chunk (async, bf16): 16 per thread
    {
      const __bf16* s = wrelp + (size_t)row * POOL + n0 + q * 16;
      __bf16* d = WR + row * G2_WS + q * 16;
      cp16(s + 0, d + 0);
      cp16(s + 8, d + 8);
    }
    if (tid < 64) BPC[tid] = bpc[n0 + tid];

    // ---- K loop: double-buffered B staging (TDM or async), 4 stages x 256 K
#if HAVE_TDM
    if (tid < 32) tdm_issue_B(wc, B, n0, 0);
#else
    {
      const __bf16* s = wc + (size_t)(n0 + row) * 1024 + q * 64;
      __bf16* d = B + row * G2_BS + q * 64;
#pragma unroll
      for (int j = 0; j < 64; j += 8) cp16(s + j, d + j);
    }
#endif
#if HAVE_TDM
    if (tid < 32) wait_tensor0();
#endif
    wait_async0();
    __syncthreads();  // stage 0 ready (U/WR also landed)

    v8f acc[2] = {};
    for (int ksi = 0; ksi < 4; ++ksi) {
      const int ks = ksi * 256;
      __bf16* Bcur = B + (ksi & 1) * 64 * G2_BS;
      if (ksi < 3) {  // issue next stage into the other buffer (overlaps WMMA)
        __bf16* Bnxt = B + ((ksi + 1) & 1) * 64 * G2_BS;
#if HAVE_TDM
        if (tid < 32) tdm_issue_B(wc, Bnxt, n0, ks + 256);
#else
        const __bf16* s = wc + (size_t)(n0 + row) * 1024 + ks + 256 + q * 64;
        __bf16* d = Bnxt + row * G2_BS + q * 64;
#pragma unroll
        for (int j = 0; j < 64; j += 8) cp16(s + j, d + j);
#endif
      }
#pragma unroll
      for (int k0 = 0; k0 < 256; k0 += 32) {
        v16bf a = ld_frag_a(A + rt * 16 * G2_AS + ks, G2_AS, m, g, k0);
#pragma unroll
        for (int c = 0; c < 2; ++c) {
          int ct = nh + 2 * c;
          v16bf b = ld_frag_b(Bcur + ct * 16 * G2_BS, G2_BS, m, g, k0);
          acc[c] = wmma_bf16(a, b, acc[c]);
        }
      }
#if HAVE_TDM
      if (tid < 32) wait_tensor0();
#endif
      wait_async0();
      __syncthreads();  // next buffer ready; everyone done with Bcur
    }

    // epilogue of GEMM2: P = (C + b_post_cat) * union -> bf16 tile in LDS
#pragma unroll
    for (int c = 0; c < 2; ++c) {
      int ct = nh + 2 * c;
      int ncl = ct * 16 + m;
      float bb = BPC[ncl];
#pragma unroll
      for (int i = 0; i < 8; ++i) {
        int rl = rt * 16 + i + 8 * g;
        float v = (acc[c][i] + bb) * U[rl * G2_US + ncl];
        P[rl * G2_PS + ncl] = (__bf16)v;
      }
    }
    __syncthreads();  // P visible to all waves

    // GEMM3 partial: relacc += P(16x64) @ WR^T
#pragma unroll
    for (int k0 = 0; k0 < 64; k0 += 32) {
      v16bf a = ld_frag_a(P + rt * 16 * G2_PS, G2_PS, m, g, k0);
#pragma unroll
      for (int c = 0; c < 2; ++c) {
        int ct = nh + 2 * c;
        v16bf b = ld_frag_b(WR + ct * 16 * G2_WS, G2_WS, m, g, k0);
        relacc[c] = wmma_bf16(a, b, relacc[c]);
      }
    }
  }

  // final: + b_rel + freq_bias[obj_preds[h], obj_preds[t]]
#pragma unroll
  for (int c = 0; c < 2; ++c) {
    int ct = nh + 2 * c;
    int cls = ct * 16 + m;
    if (cls < NRELCLS) {
      float br = brel[cls];
#pragma unroll
      for (int i = 0; i < 8; ++i) {
        int rl = rt * 16 + i + 8 * g;
        int r = r0 + rl;
        out[(size_t)r * NRELCLS + cls] = relacc[c][i] + br + freq[RB[rl] + cls];
      }
    }
  }
}

// ---------------------------------------------------------------------------
extern "C" void kernel_launch(void* const* d_in, const int* in_sizes, int n_in,
                              void* d_out, int out_size, void* d_ws, size_t ws_size,
                              hipStream_t stream) {
  (void)in_sizes; (void)n_in; (void)out_size; (void)ws_size;
  const float* edge_ctx  = (const float*)d_in[0];
  const float* unionf    = (const float*)d_in[1];
  const int*   pair_idx  = (const int*)d_in[2];
  const int*   obj_preds = (const int*)d_in[3];
  const float* W_emb     = (const float*)d_in[4];
  const float* b_emb     = (const float*)d_in[5];
  const float* W_cat     = (const float*)d_in[6];
  const float* b_cat     = (const float*)d_in[7];
  const float* W_rel     = (const float*)d_in[8];
  const float* b_rel     = (const float*)d_in[9];
  const float* freq      = (const float*)d_in[10];
  float* out = (float*)d_out;

  char* ws = (char*)d_ws;
  __bf16* head    = (__bf16*)(ws + 0);             // 16 MB
  __bf16* tail    = (__bf16*)(ws + (16u << 20));   // 16 MB
  __bf16* wc_bf   = (__bf16*)(ws + (32u << 20));   // 8 MB
  __bf16* wemb_bf = (__bf16*)(ws + (40u << 20));   // 1 MB
  __bf16* wrelp   = (__bf16*)(ws + (41u << 20));   // 512 KB

  cvt_f32_bf16<<<(1024 * 512 + 255) / 256, 256, 0, stream>>>(W_emb, wemb_bf, 1024 * 512);
  cvt_f32_bf16<<<(4096 * 1024 + 255) / 256, 256, 0, stream>>>(W_cat, wc_bf, 4096 * 1024);
  pad_wrel<<<(64 * 4096) / 256, 256, 0, stream>>>(W_rel, wrelp);

  size_t lds1 = (size_t)2 * 64 * G1_AS * sizeof(__bf16);
  gemm1_edge<<<dim3(N_OBJ / 64, 1024 / 64), 256, lds1, stream>>>(
      edge_ctx, wemb_bf, b_emb, head, tail);

  size_t lds2 = (size_t)64 * G2_AS * 2 + (size_t)2 * 64 * G2_BS * 2 +
                (size_t)64 * G2_PS * 2 + (size_t)64 * G2_WS * 2 +
                (size_t)64 * G2_US * 4 + 64 * 4 + 64 * 4 + 128 * 4;  // ~236 KB
  gemm2_fused<<<N_REL / 64, 256, lds2, stream>>>(
      head, tail, wc_bf, wrelp, unionf, b_cat, b_rel, freq, pair_idx, obj_preds, out);
}
